// Toy_model_15410342658629
// MI455X (gfx1250) — matrix operand order repaired and checked by reading, not compiled, not run on hardware
//
#include <hip/hip_runtime.h>

// ---------------- model dimensions ----------------
#define SEQ   128
#define DIMW  300
#define DIMP  30
#define INDIM 330      // DIMW + DIMP
#define INP   352      // padded to mult of 32
#define HID   300
#define G4    1200     // 4*HID
#define D     600      // 2*HID
#define DP    608      // padded to mult of 32 (and 16)
#define G3    1800     // 3*D
#define G3P   1808     // padded rows for B-operand tiles
#define NNER  5
#define NLAB  7
#define NT_D  38       // ceil(600/16)
#define NT_G4 75       // 1200/16
#define NT_G3 113      // ceil(1800/16)

typedef __attribute__((ext_vector_type(16))) _Float16 v16h;
typedef __attribute__((ext_vector_type(8)))  _Float16 v8h;
typedef __attribute__((ext_vector_type(8)))  float    v8f;

__device__ __forceinline__ float sigf(float x) { return 1.f / (1.f + __expf(-x)); }
__device__ __forceinline__ float wredsum(float v) {
#pragma unroll
    for (int o = 16; o; o >>= 1) v += __shfl_xor(v, o, 32);
    return v;
}

// ---------------- WMMA GEMM: C[m,n] = sum_k A[m,k]*W[n,k] (+bias1[n]+bias2[n]) ----------------
// A: f16 (M x lda), W: f16 (Nrp x ldw) row-major, both K-padded with zeros. One wave = one 16x16 tile.
// KP is a compile-time constant so the K-loop fully unrolls: the scheduler can clause all
// global_load_b128s, hoist them ahead of the WMMAs, and use partial loadcnt waits (software
// pipelining without rotating-register copies or branches).
template <int KP>
__global__ __launch_bounds__(32)
void wmma_gemm_nt(const _Float16* __restrict__ A, const _Float16* __restrict__ W,
                  float* __restrict__ C, int lda, int ldw, int ldc, int Nvalid,
                  const float* __restrict__ bias1, const float* __restrict__ bias2)
{
    const int lane  = threadIdx.x;
    const int mTile = blockIdx.x << 4;
    const int nTile = blockIdx.y << 4;
    const int hsel  = lane >> 4;       // 0: low half, 1: high half
    const int l16   = lane & 15;

    const _Float16* ap = A + (size_t)(mTile + l16) * lda + hsel * 16;  // A rows, K chunk kb..kb+15
    const _Float16* wp = W + (size_t)(nTile + l16) * ldw + hsel * 16;  // W rows (= B columns), K chunk kb..kb+15

    v8f acc = {0.f, 0.f, 0.f, 0.f, 0.f, 0.f, 0.f, 0.f};
#pragma unroll
    for (int k0 = 0; k0 < KP; k0 += 32) {
        v8h  alo = *(const v8h*)(ap + k0);
        v8h  ahi = *(const v8h*)(ap + k0 + 8);
        v16h b   = *(const v16h*)(wp + k0);
        v16h a;
#pragma unroll
        for (int t = 0; t < 8; ++t) { a[t] = alo[t]; a[t + 8] = ahi[t]; }
        acc = __builtin_amdgcn_wmma_f32_16x16x32_f16(false, a, false, b, (short)0, acc, false, false);
    }

    const int n = nTile + l16;
    if (n < Nvalid) {
        float bv = 0.f;
        if (bias1) bv += bias1[n];
        if (bias2) bv += bias2[n];
        const int mBase = mTile + hsel * 8;
#pragma unroll
        for (int v = 0; v < 8; ++v)
            C[(size_t)(mBase + v) * ldc + n] = acc[v] + bv;
    }
}

// ---------------- small utility kernels ----------------
__global__ void zero2_kernel(float* out) { if (threadIdx.x < 2) out[threadIdx.x] = 0.f; }

__global__ void embed_kernel(const int* __restrict__ tokens, const int* __restrict__ pos,
                             const float* __restrict__ embW, const float* __restrict__ posW,
                             _Float16* __restrict__ x16, int BN)
{
    int idx = blockIdx.x * blockDim.x + threadIdx.x;
    if (idx >= BN * INP) return;
    int m = idx / INP, c = idx % INP;
    float v = 0.f;
    if (c < DIMW)       v = embW[(size_t)tokens[m] * DIMW + c];
    else if (c < INDIM) v = posW[(size_t)pos[m] * DIMP + (c - DIMW)];
    x16[idx] = (_Float16)v;
}

__global__ void lens_kernel(const float* __restrict__ mask, int* __restrict__ lens, int B, int N)
{
    int b = threadIdx.x;
    if (b >= B) return;
    float s = 0.f;
    for (int t = 0; t < N; ++t) s += mask[b * N + t];
    lens[b] = (int)(s + 0.5f);
}

__global__ void rev16_kernel(const _Float16* __restrict__ src, _Float16* __restrict__ dst,
                             const int* __restrict__ lens, int N, int cols, int BN)
{
    int idx = blockIdx.x * blockDim.x + threadIdx.x;
    if (idx >= BN * cols) return;
    int m = idx / cols, c = idx % cols;
    int b = m / N, t = m % N;
    int L = lens[b];
    int tt = (t < L) ? (L - 1 - t) : t;
    dst[idx] = src[(size_t)(b * N + tt) * cols + c];
}

__global__ void rev32_kernel(const float* __restrict__ src, float* __restrict__ dst,
                             const int* __restrict__ lens, int N, int cols, int BN)
{
    int idx = blockIdx.x * blockDim.x + threadIdx.x;
    if (idx >= BN * cols) return;
    int m = idx / cols, c = idx % cols;
    int b = m / N, t = m % N;
    int L = lens[b];
    int tt = (t < L) ? (L - 1 - t) : t;
    dst[idx] = src[(size_t)(b * N + tt) * cols + c];
}

// f32 (Nr x srs, col offset sco) -> zero-padded f16 (Nrp x Kp)
__global__ void cvt_w_kernel(const float* __restrict__ src, _Float16* __restrict__ dst,
                             int Nr, int Kc, int Kp, int srs, int sco, int total)
{
    int idx = blockIdx.x * blockDim.x + threadIdx.x;
    if (idx >= total) return;
    int r = idx / Kp, c = idx % Kp;
    float v = (r < Nr && c < Kc) ? src[(size_t)r * srs + sco + c] : 0.f;
    dst[idx] = (_Float16)v;
}

// ---------------- LSTM scan (one block per batch row, sequential over time) ----------------
__global__ __launch_bounds__(512)
void lstm_scan_kernel(const float* __restrict__ gi,  // (B,N,4H) = x@Wih.T + bih + bhh
                      const float* __restrict__ Whh, // (4H,H) f32 row-major
                      float* __restrict__ hout, int N)
{
    __shared__ float h_s[HID], c_s[HID], g_s[G4];
    const int b = blockIdx.x, tid = threadIdx.x;
    for (int d = tid; d < HID; d += blockDim.x) { h_s[d] = 0.f; c_s[d] = 0.f; }
    __syncthreads();
    const float* gib = gi + (size_t)b * N * G4;
    float* hb = hout + (size_t)b * N * HID;
    for (int t = 0; t < N; ++t) {
        const float* git = gib + (size_t)t * G4;
        for (int col = tid; col < G4; col += blockDim.x) {
            const float* wr = Whh + (size_t)col * HID;
            float s = git[col];
            for (int k = 0; k < HID; ++k) s += h_s[k] * wr[k];
            g_s[col] = s;
        }
        __syncthreads();
        if (tid < HID) {
            float ig = sigf(g_s[tid]);
            float fg = sigf(g_s[HID + tid]);
            float gg = tanhf(g_s[2 * HID + tid]);
            float og = sigf(g_s[3 * HID + tid]);
            float c = fg * c_s[tid] + ig * gg;
            float h = og * tanhf(c);
            c_s[tid] = c; h_s[tid] = h;
            hb[(size_t)t * HID + tid] = h;
        }
        __syncthreads();
    }
}

// ---------------- build H = concat(hf,hb)*mask; init Hg,Hr,H16,Hr16 ----------------
__global__ void buildH_kernel(const float* __restrict__ hf, const float* __restrict__ hb,
                              const float* __restrict__ mask,
                              float* __restrict__ H, float* __restrict__ Hg, float* __restrict__ Hr,
                              _Float16* __restrict__ H16, _Float16* __restrict__ Hr16, int BN)
{
    int idx = blockIdx.x * blockDim.x + threadIdx.x;
    if (idx >= BN * DP) return;
    int m = idx / DP, dp = idx % DP;
    float v = 0.f;
    if (dp < HID)    v = hf[(size_t)m * HID + dp] * mask[m];
    else if (dp < D) v = hb[(size_t)m * HID + (dp - HID)] * mask[m];
    if (dp < D) {
        size_t o = (size_t)m * D + dp;
        H[o] = v; Hg[o] = v; Hr[o] = v;
    }
    H16[idx]  = (_Float16)v;
    Hr16[idx] = (_Float16)v;
}

// ---------------- Cg head: logits / softmax / final loss+pred ----------------
__global__ __launch_bounds__(32)
void cg_kernel(const float* __restrict__ Hg, const float* __restrict__ CgW, const float* __restrict__ Cgb,
               const float* __restrict__ mask, const int* __restrict__ labels,
               float* __restrict__ prob_out, float* __restrict__ pred_out, float* __restrict__ loss_out,
               int mode, float invB)
{
    const int m = blockIdx.x, lane = threadIdx.x;
    const float* hr = Hg + (size_t)m * D;
    float acc[NNER] = {0.f, 0.f, 0.f, 0.f, 0.f};
    for (int k = lane; k < D; k += 32) {
        float hv = hr[k];
#pragma unroll
        for (int l = 0; l < NNER; ++l) acc[l] += hv * CgW[l * D + k];
    }
#pragma unroll
    for (int l = 0; l < NNER; ++l) acc[l] = wredsum(acc[l]);
    if (lane == 0) {
        float lg[NNER], mx = -1e30f;
#pragma unroll
        for (int l = 0; l < NNER; ++l) { lg[l] = acc[l] + Cgb[l]; mx = fmaxf(mx, lg[l]); }
        float ex[NNER], s = 0.f;
#pragma unroll
        for (int l = 0; l < NNER; ++l) { ex[l] = __expf(lg[l] - mx); s += ex[l]; }
        if (mode == 0) {
#pragma unroll
            for (int l = 0; l < NNER; ++l) prob_out[(size_t)m * NNER + l] = ex[l] / s;
        } else {
            int arg = 0;
#pragma unroll
            for (int l = 1; l < NNER; ++l) if (lg[l] > lg[arg]) arg = l;
            float mval = mask[m];
            pred_out[m] = (float)arg * mval;
            float lse = mx + __logf(s);
            int y = labels[m];
            float ly = lg[0];
#pragma unroll
            for (int l = 1; l < NNER; ++l) if (l == y) ly = lg[l];
            atomicAdd(loss_out, -(ly - lse) * mval * invB);
        }
    }
}

// ---------------- pairwise: max over j of sigmoid(relu(A_i + C_j) . CrW_l + b) * mask ----------------
__global__ __launch_bounds__(256)
void pairwise_max_kernel(const float* __restrict__ Ar, const float* __restrict__ Cc,
                         const float* __restrict__ CrW, const float* __restrict__ Crb,
                         const float* __restrict__ mask, float* __restrict__ probCr, int N)
{
    __shared__ float a_s[D];
    __shared__ float w_s[NLAB * D];
    __shared__ float red[8 * NLAB];
    const int blk = blockIdx.x;     // b*N + i
    const int tid = threadIdx.x, lane = tid & 31, wv = tid >> 5;
    const float* arow = Ar + (size_t)blk * D;
    for (int k = tid; k < D; k += 256)        a_s[k] = arow[k];
    for (int k = tid; k < NLAB * D; k += 256) w_s[k] = CrW[k];
    __syncthreads();
    const int b = blk / N;
    const float mi = mask[blk];
    float cb[NLAB];
#pragma unroll
    for (int l = 0; l < NLAB; ++l) cb[l] = Crb[l];
    float mx[NLAB] = {0.f, 0.f, 0.f, 0.f, 0.f, 0.f, 0.f};
    for (int j = wv; j < N; j += 8) {
        const float* crow = Cc + (size_t)(b * N + j) * D;
        float acc[NLAB] = {0.f, 0.f, 0.f, 0.f, 0.f, 0.f, 0.f};
        for (int k = lane; k < D; k += 32) {
            float e = a_s[k] + crow[k];
            e = e > 0.f ? e : 0.f;
#pragma unroll
            for (int l = 0; l < NLAB; ++l) acc[l] += e * w_s[l * D + k];
        }
#pragma unroll
        for (int l = 0; l < NLAB; ++l) acc[l] = wredsum(acc[l]);
        float mm = mi * mask[b * N + j];
#pragma unroll
        for (int l = 0; l < NLAB; ++l) mx[l] = fmaxf(mx[l], mm * sigf(acc[l] + cb[l]));
    }
    if (lane == 0) {
#pragma unroll
        for (int l = 0; l < NLAB; ++l) red[wv * NLAB + l] = mx[l];
    }
    __syncthreads();
    if (tid < NLAB) {
        float m0 = 0.f;
#pragma unroll
        for (int w = 0; w < 8; ++w) m0 = fmaxf(m0, red[w * NLAB + tid]);
        probCr[(size_t)blk * NLAB + tid] = m0;
    }
}

// ---------------- final pairwise: rc_pred + bce loss ----------------
__global__ __launch_bounds__(256)
void pairwise_final_kernel(const float* __restrict__ Ar, const float* __restrict__ Cc,
                           const float* __restrict__ CrW, const float* __restrict__ Crb,
                           const float* __restrict__ mask, const float* __restrict__ RC,
                           float* __restrict__ rc_pred, float* __restrict__ rc_loss,
                           int N, float invB)
{
    __shared__ float a_s[D];
    __shared__ float w_s[NLAB * D];
    __shared__ float rsum[8];
    const int blk = blockIdx.x;
    const int tid = threadIdx.x, lane = tid & 31, wv = tid >> 5;
    const float* arow = Ar + (size_t)blk * D;
    for (int k = tid; k < D; k += 256)        a_s[k] = arow[k];
    for (int k = tid; k < NLAB * D; k += 256) w_s[k] = CrW[k];
    __syncthreads();
    const int b = blk / N;
    const float mi = mask[blk];
    float cb[NLAB];
#pragma unroll
    for (int l = 0; l < NLAB; ++l) cb[l] = Crb[l];
    float bacc = 0.f;
    for (int j = wv; j < N; j += 8) {
        const float* crow = Cc + (size_t)(b * N + j) * D;
        float acc[NLAB] = {0.f, 0.f, 0.f, 0.f, 0.f, 0.f, 0.f};
        for (int k = lane; k < D; k += 32) {
            float e = a_s[k] + crow[k];
            e = e > 0.f ? e : 0.f;
#pragma unroll
            for (int l = 0; l < NLAB; ++l) acc[l] += e * w_s[l * D + k];
        }
#pragma unroll
        for (int l = 0; l < NLAB; ++l) acc[l] = wredsum(acc[l]);
        float mm = mi * mask[b * N + j];
        if (lane < NLAB) {
            float av = acc[0] + cb[0];
#pragma unroll
            for (int l = 1; l < NLAB; ++l) if (lane == l) av = acc[l] + cb[l];
            float s = sigf(av);
            size_t o = ((size_t)blk * N + j) * NLAB + lane;
            rc_pred[o] = s * mm;
            float pr = fminf(fmaxf(s, 1e-7f), 1.f - 1e-7f);
            float y = RC[o];
            float bce = -(y * __logf(pr) + (1.f - y) * log1pf(-pr));
            bacc += bce * mm;
        }
    }
    bacc = wredsum(bacc);
    if (lane == 0) rsum[wv] = bacc;
    __syncthreads();
    if (tid == 0) {
        float s = 0.f;
#pragma unroll
        for (int w = 0; w < 8; ++w) s += rsum[w];
        atomicAdd(rc_loss, s * invB);
    }
}

// ---------------- GRU combine: Hout = (1-z)*n + z*H ----------------
__global__ void gru_combine_kernel(const float* __restrict__ prob, int P,
                                   const float* __restrict__ Wih, const float* __restrict__ bih,
                                   const float* __restrict__ gh,  // (BN,1800) includes bhh
                                   const float* __restrict__ H, float* __restrict__ Hout, int total)
{
    int idx = blockIdx.x * blockDim.x + threadIdx.x;
    if (idx >= total) return;
    int m = idx / D, d = idx % D;
    const float* pr = prob + (size_t)m * P;
    float gi[3];
#pragma unroll
    for (int g = 0; g < 3; ++g) {
        int col = g * D + d;
        float s = bih[col];
        const float* wr = Wih + (size_t)col * P;
        for (int p = 0; p < P; ++p) s += pr[p] * wr[p];
        gi[g] = s;
    }
    const float* ghr = gh + (size_t)m * G3;
    float r = sigf(gi[0] + ghr[d]);
    float z = sigf(gi[1] + ghr[D + d]);
    float n = tanhf(gi[2] + r * ghr[2 * D + d]);
    Hout[idx] = (1.f - z) * n + z * H[idx];
}

// ---------------- H += Hr + Hg, refresh f16 copies ----------------
__global__ void h_update_kernel(float* __restrict__ H, const float* __restrict__ Hr,
                                const float* __restrict__ Hg,
                                _Float16* __restrict__ H16, _Float16* __restrict__ Hr16, int BN)
{
    int idx = blockIdx.x * blockDim.x + threadIdx.x;
    if (idx >= BN * DP) return;
    int m = idx / DP, dp = idx % DP;
    if (dp < D) {
        size_t o = (size_t)m * D + dp;
        float nh = H[o] + Hr[o] + Hg[o];
        H[o] = nh;
        H16[idx]  = (_Float16)nh;
        Hr16[idx] = (_Float16)Hr[o];
    } else {
        H16[idx]  = (_Float16)0.f;
        Hr16[idx] = (_Float16)0.f;
    }
}

// ---------------- host: orchestration ----------------
extern "C" void kernel_launch(void* const* d_in, const int* in_sizes, int n_in,
                              void* d_out, int out_size, void* d_ws, size_t ws_size,
                              hipStream_t stream)
{
    const int* tokens = (const int*)d_in[0];
    const int* posi   = (const int*)d_in[1];
    const float* mask = (const float*)d_in[2];
    const int* NERl   = (const int*)d_in[3];
    const float* RCl  = (const float*)d_in[4];
    const float* embW = (const float*)d_in[5];
    const float* posW = (const float*)d_in[6];
    const float* Wih_f = (const float*)d_in[7];
    const float* Whh_f = (const float*)d_in[8];
    const float* bih_f = (const float*)d_in[9];
    const float* bhh_f = (const float*)d_in[10];
    const float* Wih_b = (const float*)d_in[11];
    const float* Whh_b = (const float*)d_in[12];
    const float* bih_b = (const float*)d_in[13];
    const float* bhh_b = (const float*)d_in[14];
    const float* GgWih = (const float*)d_in[15];
    const float* GgWhh = (const float*)d_in[16];
    const float* Ggbih = (const float*)d_in[17];
    const float* Ggbhh = (const float*)d_in[18];
    const float* GrWih = (const float*)d_in[19];
    const float* GrWhh = (const float*)d_in[20];
    const float* Grbih = (const float*)d_in[21];
    const float* Grbhh = (const float*)d_in[22];
    const float* LrW   = (const float*)d_in[23];
    const float* Lrb   = (const float*)d_in[24];
    const float* CrW   = (const float*)d_in[25];
    const float* Crb   = (const float*)d_in[26];
    const float* CgW   = (const float*)d_in[27];
    const float* Cgb   = (const float*)d_in[28];

    float* out = (float*)d_out;
    const int BN = in_sizes[0];
    const int Bc = BN / SEQ;
    const float invB = 1.0f / (float)Bc;

    // bump allocator on d_ws
    size_t off = 0;
    auto alloc = [&](size_t bytes) -> void* {
        void* p = (char*)d_ws + off;
        off = (off + bytes + 255) & ~(size_t)255;
        return p;
    };
    _Float16* x16    = (_Float16*)alloc((size_t)BN * INP * 2);
    _Float16* xrev16 = (_Float16*)alloc((size_t)BN * INP * 2);
    _Float16* Wf16   = (_Float16*)alloc((size_t)G4 * INP * 2);
    _Float16* Wb16   = (_Float16*)alloc((size_t)G4 * INP * 2);
    _Float16* W1h    = (_Float16*)alloc((size_t)DP * DP * 2);
    _Float16* W2h    = (_Float16*)alloc((size_t)DP * DP * 2);
    _Float16* Gg16   = (_Float16*)alloc((size_t)G3P * DP * 2);
    _Float16* Gr16   = (_Float16*)alloc((size_t)G3P * DP * 2);
    float* gi_f  = (float*)alloc((size_t)BN * G4 * 4);
    float* gi_b  = (float*)alloc((size_t)BN * G4 * 4);
    float* hf    = (float*)alloc((size_t)BN * HID * 4);
    float* hbrev = (float*)alloc((size_t)BN * HID * 4);
    float* hbv   = (float*)alloc((size_t)BN * HID * 4);
    float* H     = (float*)alloc((size_t)BN * D * 4);
    float* Hg    = (float*)alloc((size_t)BN * D * 4);
    float* Hr    = (float*)alloc((size_t)BN * D * 4);
    _Float16* H16  = (_Float16*)alloc((size_t)BN * DP * 2);
    _Float16* Hr16 = (_Float16*)alloc((size_t)BN * DP * 2);
    float* Apair  = (float*)alloc((size_t)BN * D * 4);
    float* Cpair  = (float*)alloc((size_t)BN * D * 4);
    float* gh     = (float*)alloc((size_t)BN * G3 * 4);
    float* probCg = (float*)alloc((size_t)BN * NNER * 4);
    float* probCr = (float*)alloc((size_t)BN * NLAB * 4);
    int*   lens   = (int*)alloc((size_t)Bc * 4);

    const int TB = 256;
    auto nblk = [](long long n, int t) { return (unsigned)((n + t - 1) / t); };

    zero2_kernel<<<1, 32, 0, stream>>>(out);

    // embeddings -> f16 (padded), lens, reversed x
    embed_kernel<<<nblk((long long)BN * INP, TB), TB, 0, stream>>>(tokens, posi, embW, posW, x16, BN);
    lens_kernel<<<1, 64, 0, stream>>>(mask, lens, Bc, SEQ);
    rev16_kernel<<<nblk((long long)BN * INP, TB), TB, 0, stream>>>(x16, xrev16, lens, SEQ, INP, BN);

    // weight conversions to padded f16
    cvt_w_kernel<<<nblk((long long)G4 * INP, TB), TB, 0, stream>>>(Wih_f, Wf16, G4, INDIM, INP, INDIM, 0, G4 * INP);
    cvt_w_kernel<<<nblk((long long)G4 * INP, TB), TB, 0, stream>>>(Wih_b, Wb16, G4, INDIM, INP, INDIM, 0, G4 * INP);
    cvt_w_kernel<<<nblk((long long)DP * DP, TB), TB, 0, stream>>>(LrW, W1h, D, D, DP, 2 * D, 0, DP * DP);
    cvt_w_kernel<<<nblk((long long)DP * DP, TB), TB, 0, stream>>>(LrW, W2h, D, D, DP, 2 * D, D, DP * DP);
    cvt_w_kernel<<<nblk((long long)G3P * DP, TB), TB, 0, stream>>>(GgWhh, Gg16, G3, D, DP, D, 0, G3P * DP);
    cvt_w_kernel<<<nblk((long long)G3P * DP, TB), TB, 0, stream>>>(GrWhh, Gr16, G3, D, DP, D, 0, G3P * DP);

    // LSTM input projections (WMMA, K fully unrolled) with biases folded in
    wmma_gemm_nt<INP><<<dim3(BN / 16, NT_G4), 32, 0, stream>>>(x16,    Wf16, gi_f, INP, INP, G4, G4, bih_f, bhh_f);
    wmma_gemm_nt<INP><<<dim3(BN / 16, NT_G4), 32, 0, stream>>>(xrev16, Wb16, gi_b, INP, INP, G4, G4, bih_b, bhh_b);

    // recurrent scans
    lstm_scan_kernel<<<Bc, 512, 0, stream>>>(gi_f, Whh_f, hf, SEQ);
    lstm_scan_kernel<<<Bc, 512, 0, stream>>>(gi_b, Whh_b, hbrev, SEQ);
    rev32_kernel<<<nblk((long long)BN * HID, TB), TB, 0, stream>>>(hbrev, hbv, lens, SEQ, HID, BN);

    buildH_kernel<<<nblk((long long)BN * DP, TB), TB, 0, stream>>>(hf, hbv, mask, H, Hg, Hr, H16, Hr16, BN);

    for (int r = 0; r < 2; ++r) {
        cg_kernel<<<BN, 32, 0, stream>>>(Hg, CgW, Cgb, mask, NERl, probCg, nullptr, nullptr, 0, invB);
        wmma_gemm_nt<DP><<<dim3(BN / 16, NT_D), 32, 0, stream>>>(Hr16, W1h, Apair, DP, DP, D, D, Lrb, nullptr);
        wmma_gemm_nt<DP><<<dim3(BN / 16, NT_D), 32, 0, stream>>>(Hr16, W2h, Cpair, DP, DP, D, D, nullptr, nullptr);
        pairwise_max_kernel<<<BN, 256, 0, stream>>>(Apair, Cpair, CrW, Crb, mask, probCr, SEQ);

        wmma_gemm_nt<DP><<<dim3(BN / 16, NT_G3), 32, 0, stream>>>(H16, Gg16, gh, DP, DP, G3, G3, Ggbhh, nullptr);
        gru_combine_kernel<<<nblk((long long)BN * D, TB), TB, 0, stream>>>(probCg, NNER, GgWih, Ggbih, gh, H, Hg, BN * D);
        wmma_gemm_nt<DP><<<dim3(BN / 16, NT_G3), 32, 0, stream>>>(H16, Gr16, gh, DP, DP, G3, G3, Grbhh, nullptr);
        gru_combine_kernel<<<nblk((long long)BN * D, TB), TB, 0, stream>>>(probCr, NLAB, GrWih, Grbih, gh, H, Hr, BN * D);

        h_update_kernel<<<nblk((long long)BN * DP, TB), TB, 0, stream>>>(H, Hr, Hg, H16, Hr16, BN);
    }

    // final NER head: pred + loss
    cg_kernel<<<BN, 32, 0, stream>>>(Hg, CgW, Cgb, mask, NERl, nullptr, out + 2, out + 0, 1, invB);

    // final pairwise RC head: full predictions + bce loss
    wmma_gemm_nt<DP><<<dim3(BN / 16, NT_D), 32, 0, stream>>>(Hr16, W1h, Apair, DP, DP, D, D, Lrb, nullptr);
    wmma_gemm_nt<DP><<<dim3(BN / 16, NT_D), 32, 0, stream>>>(Hr16, W2h, Cpair, DP, DP, D, D, nullptr, nullptr);
    pairwise_final_kernel<<<BN, 256, 0, stream>>>(Apair, Cpair, CrW, Crb, mask, RCl,
                                                  out + 2 + BN, out + 1, SEQ, invB);
    (void)n_in; (void)out_size; (void)ws_size;
}
